// CasualSelfAttention_1348619731345
// MI455X (gfx1250) — compile-verified, matrix operand order checked by reading, not run on hardware
//
#include <hip/hip_runtime.h>
#include <hip/hip_bf16.h>
#include <math.h>

// ---------------------------------------------------------------------------
// Causal self-attention (B=2, T=2048, C=1024, H=16, HD=64) for gfx1250.
// Matrix work on v_wmma_f32_16x16x32_f16; staging via async LDS copies
// (ASYNCcnt) with double buffering.
// ---------------------------------------------------------------------------

typedef _Float16 half_t;
typedef __attribute__((ext_vector_type(8)))  _Float16 v8h;
typedef __attribute__((ext_vector_type(16))) _Float16 v16h;
typedef __attribute__((ext_vector_type(8)))  float    v8f;
typedef __attribute__((ext_vector_type(4)))  int      v4i_t;

static constexpr int Bb  = 2;
static constexpr int Tt  = 2048;
static constexpr int Cc  = 1024;
static constexpr int Hh  = 16;
static constexpr int HDd = 64;
static constexpr int C3  = 3 * Cc;

__device__ __forceinline__ v16h cat16(v8h lo, v8h hi) {
  return __builtin_shufflevector(lo, hi, 0,1,2,3,4,5,6,7,8,9,10,11,12,13,14,15);
}
__device__ __forceinline__ v8f zero8f() {
  v8f z = {0.f,0.f,0.f,0.f,0.f,0.f,0.f,0.f};
  return z;
}
__device__ __forceinline__ v8f wmma16(v16h a, v16h b, v8f c) {
  return __builtin_amdgcn_wmma_f32_16x16x32_f16(false, a, false, b, (short)0, c,
                                                false, false);
}

// 16-byte global -> LDS copy. Async (ASYNCcnt-tracked) when available.
__device__ __forceinline__ void cp16(half_t* lds_dst, const half_t* g_src) {
#if __has_builtin(__builtin_amdgcn_global_load_async_to_lds_b128)
  __builtin_amdgcn_global_load_async_to_lds_b128(
      (__attribute__((address_space(1))) v4i_t*)g_src,
      (__attribute__((address_space(3))) v4i_t*)lds_dst,
      0, 0);
#else
  *(v8h*)lds_dst = *(const v8h*)g_src;
#endif
}
__device__ __forceinline__ void wait_async0() {
#if __has_builtin(__builtin_amdgcn_s_wait_asynccnt)
  __builtin_amdgcn_s_wait_asynccnt(0);
#else
  asm volatile("s_wait_asynccnt 0" ::: "memory");
#endif
}

// ---------------------------------------------------------------------------
// Elementwise staging kernels
// ---------------------------------------------------------------------------
__global__ void cvt_f32_to_f16(const float* __restrict__ src,
                               half_t* __restrict__ dst, int n) {
  int i = blockIdx.x * blockDim.x + threadIdx.x;
  if (i < n) dst[i] = (half_t)src[i];
}

// src [R][Ncol] f32 row-major -> dst [Ncol][R] f16 (transposed)
__global__ void transpose_to_f16(const float* __restrict__ src,
                                 half_t* __restrict__ dst, int R, int Ncol) {
  int i = blockIdx.x * blockDim.x + threadIdx.x;
  if (i >= R * Ncol) return;
  int row = i / Ncol;
  int col = i - row * Ncol;
  dst[(size_t)col * R + row] = (half_t)src[i];
}

// vt[b,h,d,t] = qkv[b,t, 2C + h*64 + d]
__global__ void v_transpose(const half_t* __restrict__ qkv,
                            half_t* __restrict__ vt) {
  int i = blockIdx.x * blockDim.x + threadIdx.x;   // ((b*H+h)*64+d)*T + t
  if (i >= Bb * Hh * HDd * Tt) return;
  int t    = i & (Tt - 1);
  int rest = i >> 11;
  int d    = rest & (HDd - 1);
  rest   >>= 6;
  int h    = rest & (Hh - 1);
  int b    = rest >> 4;
  vt[i] = qkv[(size_t)(b * Tt + t) * C3 + 2 * Cc + h * HDd + d];
}

// ---------------------------------------------------------------------------
// WMMA GEMM:  C[M,N] = A[M,K] * Bt[N,K]^T + bias[N]
// 64x64 tile, K-step 32, 4 waves, double-buffered async LDS staging.
// ---------------------------------------------------------------------------
template <bool HALF_OUT>
__global__ __launch_bounds__(128)
void gemm_wmma(const half_t* __restrict__ A, const half_t* __restrict__ Bt,
               const float* __restrict__ bias, void* __restrict__ Cout,
               int M, int N, int K) {
  (void)M;
  constexpr int LDSP = 40;   // padded row stride (f16): 80B, 16B-aligned chunks
  __shared__ __align__(16) half_t As[2][64 * LDSP];
  __shared__ __align__(16) half_t Bs[2][64 * LDSP];

  const int tid  = threadIdx.x;
  const int wave = tid >> 5;
  const int lane = tid & 31;
  const int r    = lane & 15;
  const int hf   = lane >> 4;
  const int m0   = blockIdx.y * 64;
  const int n0   = blockIdx.x * 64;

  auto stage = [&](int k0, int buf) {
    #pragma unroll
    for (int i = 0; i < 2; ++i) {
      int chunk = tid + i * 128;         // 256 16B-chunks per 64x32 tile
      int row   = chunk >> 2;
      int koff  = (chunk & 3) * 8;
      cp16(&As[buf][row * LDSP + koff],
           &A[(size_t)(m0 + row) * K + k0 + koff]);
      cp16(&Bs[buf][row * LDSP + koff],
           &Bt[(size_t)(n0 + row) * K + k0 + koff]);
    }
  };

  v8f acc[4] = {zero8f(), zero8f(), zero8f(), zero8f()};

  stage(0, 0);
  const int nsteps = K >> 5;
  for (int it = 0; it < nsteps; ++it) {
    const int cur = it & 1;
    wait_async0();
    __syncthreads();                       // tile `it` visible; prev reads done
    if (it + 1 < nsteps) stage((it + 1) << 5, cur ^ 1);  // overlap with math

    const int arow = wave * 16 + r;
    v8h alo = *(const v8h*)&As[cur][arow * LDSP + hf * 8];
    v8h ahi = *(const v8h*)&As[cur][arow * LDSP + 16 + hf * 8];
    v16h af = cat16(alo, ahi);

    #pragma unroll
    for (int c = 0; c < 4; ++c) {
      const int brow = c * 16 + r;
      v8h blo = *(const v8h*)&Bs[cur][brow * LDSP + hf * 8];
      v8h bhi = *(const v8h*)&Bs[cur][brow * LDSP + 16 + hf * 8];
      acc[c]  = wmma16(af, cat16(blo, bhi), acc[c]);
    }
  }

  const int mrow = m0 + wave * 16;
  #pragma unroll
  for (int c = 0; c < 4; ++c) {
    const int col = n0 + c * 16 + r;
    const float bv = bias[col];
    #pragma unroll
    for (int j = 0; j < 8; ++j) {
      const int row = mrow + j + hf * 8;
      const float v = acc[c][j] + bv;
      if (HALF_OUT)
        ((half_t*)Cout)[(size_t)row * N + col] = (half_t)v;
      else
        ((float*)Cout)[(size_t)row * N + col] = v;
    }
  }
}

// ---------------------------------------------------------------------------
// Flash attention: block = 64 queries (4 waves x 16), shared K/V tiles staged
// through LDS with async copies + double buffering. 32 keys per iteration:
// 4 score WMMAs + 4 PV WMMAs per wave. Online softmax via half-wave shuffles;
// per-wave LDS bounce (s_wait_dscnt fence) relayouts P into an A fragment.
// ---------------------------------------------------------------------------
__global__ __launch_bounds__(128)
void attn_flash(const half_t* __restrict__ qkv, const half_t* __restrict__ vt,
                half_t* __restrict__ y) {
  constexpr int KP = 72;   // K-tile row stride (halfs): 144B, conflict-free
  constexpr int VP = 40;   // V-tile row stride (halfs): 80B, conflict-free
  __shared__ __align__(16) half_t Ks[2][32 * KP];
  __shared__ __align__(16) half_t Vs[2][64 * VP];
  __shared__ __align__(16) half_t plds[4][16 * 32];

  const int tid  = threadIdx.x;
  const int wave = tid >> 5;
  const int lane = tid & 31;
  const int r    = lane & 15;
  const int hf   = lane >> 4;

  const int bh  = blockIdx.y;            // b*H + h
  const int b   = bh >> 4;
  const int h   = bh & (Hh - 1);
  const int q0b = blockIdx.x * 64;       // block query base
  const int q0  = q0b + wave * 16;       // this wave's query tile

  const half_t* qbase = qkv + (size_t)(b * Tt) * C3 + h * HDd;
  const half_t* kbase = qbase + Cc;
  const half_t* vbase = vt + (size_t)bh * HDd * Tt;

  auto stage = [&](int kb, int buf) {
    const int n0 = kb * 32;
    #pragma unroll
    for (int i = 0; i < 2; ++i) {
      int c = tid + i * 128;
      // K tile: 32 rows x 64 halfs = 256 chunks
      int krow = c >> 3, kof = (c & 7) * 8;
      cp16(&Ks[buf][krow * KP + kof],
           kbase + (size_t)(n0 + krow) * C3 + kof);
      // V tile: 64 rows (d) x 32 halfs = 256 chunks
      int vrow = c >> 2, vof = (c & 3) * 8;
      cp16(&Vs[buf][vrow * VP + vof],
           vbase + (size_t)vrow * Tt + n0 + vof);
    }
  };

  // Q fragments (16 x 64 -> two 16x32 A fragments), resident all kernel
  v16h qf[2];
  #pragma unroll
  for (int s = 0; s < 2; ++s) {
    const half_t* p = qbase + (size_t)(q0 + r) * C3 + s * 32 + hf * 8;
    qf[s] = cat16(*(const v8h*)p, *(const v8h*)(p + 16));
  }

  v8f o[4] = {zero8f(), zero8f(), zero8f(), zero8f()};
  float mrow[8], lrow[8];
  #pragma unroll
  for (int j = 0; j < 8; ++j) { mrow[j] = -1e30f; lrow[j] = 0.f; }

  const int kbmax = (q0b + 63) >> 5;     // uniform across the block
  stage(0, 0);
  for (int kb = 0; kb <= kbmax; ++kb) {
    const int cur = kb & 1;
    const int n0  = kb * 32;
    wait_async0();
    __syncthreads();                     // tile kb visible; prev reads done
    if (kb + 1 <= kbmax) stage(kb + 1, cur ^ 1);

    // ---- S = Q*K^T (16 queries x 32 keys) ----
    v8f sc[2] = {zero8f(), zero8f()};
    #pragma unroll
    for (int cc = 0; cc < 2; ++cc) {
      const half_t* kp = &Ks[cur][(cc * 16 + r) * KP];
      v16h kf0 = cat16(*(const v8h*)(kp + hf * 8),
                       *(const v8h*)(kp + 16 + hf * 8));
      v16h kf1 = cat16(*(const v8h*)(kp + 32 + hf * 8),
                       *(const v8h*)(kp + 48 + hf * 8));
      sc[cc] = wmma16(qf[0], kf0, sc[cc]);
      sc[cc] = wmma16(qf[1], kf1, sc[cc]);
    }

    const bool need_mask = (n0 + 31) > q0;

    // ---- online softmax, rows live in (VGPR j, half-wave) ----
    #pragma unroll
    for (int j = 0; j < 8; ++j) {
      const int qrow = q0 + j + hf * 8;
      float s0 = sc[0][j] * 0.125f;      // 1/sqrt(64)
      float s1 = sc[1][j] * 0.125f;
      if (need_mask) {
        if (n0 + r > qrow)      s0 = -1e30f;
        if (n0 + 16 + r > qrow) s1 = -1e30f;
      }
      float mx = fmaxf(s0, s1);
      #pragma unroll
      for (int off = 1; off < 16; off <<= 1)
        mx = fmaxf(mx, __shfl_xor(mx, off, 16));
      const float mold = mrow[j];
      const float mnew = fmaxf(mold, mx);
      const float corr = __expf(mold - mnew);
      const float p0   = __expf(s0 - mnew);
      const float p1   = __expf(s1 - mnew);
      float rs = p0 + p1;
      #pragma unroll
      for (int off = 1; off < 16; off <<= 1)
        rs += __shfl_xor(rs, off, 16);
      mrow[j] = mnew;
      lrow[j] = lrow[j] * corr + rs;
      o[0][j] *= corr; o[1][j] *= corr; o[2][j] *= corr; o[3][j] *= corr;

      half_t* pl = &plds[wave][(j + hf * 8) * 32];
      pl[r]      = (half_t)p0;
      pl[16 + r] = (half_t)p1;
    }

    // same-wave DS ordering fence (split counter) -- no block barrier needed
    asm volatile("s_wait_dscnt 0" ::: "memory");

    const half_t* pb = &plds[wave][0];
    v16h pf = cat16(*(const v8h*)(pb + r * 32 + hf * 8),
                    *(const v8h*)(pb + r * 32 + 16 + hf * 8));

    #pragma unroll
    for (int dt = 0; dt < 4; ++dt) {
      const half_t* vp = &Vs[cur][(dt * 16 + r) * VP + hf * 8];
      o[dt] = wmma16(pf, cat16(*(const v8h*)vp, *(const v8h*)(vp + 16)), o[dt]);
    }
  }

  #pragma unroll
  for (int j = 0; j < 8; ++j) {
    const int row = q0 + j + hf * 8;
    const float inv = 1.0f / lrow[j];
    #pragma unroll
    for (int dt = 0; dt < 4; ++dt) {
      y[(size_t)(b * Tt + row) * Cc + h * HDd + dt * 16 + r] =
          (half_t)(o[dt][j] * inv);
    }
  }
}

// ---------------------------------------------------------------------------
// Host orchestration
// ---------------------------------------------------------------------------
extern "C" void kernel_launch(void* const* d_in, const int* in_sizes, int n_in,
                              void* d_out, int out_size, void* d_ws, size_t ws_size,
                              hipStream_t stream) {
  (void)in_sizes; (void)n_in; (void)out_size; (void)ws_size;

  const float* x      = (const float*)d_in[0];
  const float* w_attn = (const float*)d_in[1];
  const float* b_attn = (const float*)d_in[2];
  const float* w_proj = (const float*)d_in[3];
  const float* b_proj = (const float*)d_in[4];
  float* out = (float*)d_out;

  const int M = Bb * Tt;                  // 4096
  char* ws = (char*)d_ws;
  size_t off = 0;
  half_t* x_h  = (half_t*)(ws + off); off += (size_t)M * Cc * 2;
  half_t* wA_t = (half_t*)(ws + off); off += (size_t)C3 * Cc * 2;
  half_t* wP_t = (half_t*)(ws + off); off += (size_t)Cc * Cc * 2;
  half_t* qkv  = (half_t*)(ws + off); off += (size_t)M * C3 * 2;
  half_t* vt   = (half_t*)(ws + off); off += (size_t)Bb * Hh * HDd * Tt * 2;
  half_t* y_h  = (half_t*)(ws + off); off += (size_t)M * Cc * 2;

  {
    int n = M * Cc;
    cvt_f32_to_f16<<<(n + 255) / 256, 256, 0, stream>>>(x, x_h, n);
  }
  {
    int n = Cc * C3;
    transpose_to_f16<<<(n + 255) / 256, 256, 0, stream>>>(w_attn, wA_t, Cc, C3);
  }
  {
    int n = Cc * Cc;
    transpose_to_f16<<<(n + 255) / 256, 256, 0, stream>>>(w_proj, wP_t, Cc, Cc);
  }
  {
    dim3 grid(C3 / 64, M / 64);
    gemm_wmma<true><<<grid, 128, 0, stream>>>(x_h, wA_t, b_attn, qkv, M, C3, Cc);
  }
  {
    int n = Bb * Hh * HDd * Tt;
    v_transpose<<<(n + 255) / 256, 256, 0, stream>>>(qkv, vt);
  }
  {
    dim3 grid(Tt / 64, Bb * Hh);
    attn_flash<<<grid, 128, 0, stream>>>(qkv, vt, y_h);
  }
  {
    dim3 grid(Cc / 64, M / 64);
    gemm_wmma<false><<<grid, 128, 0, stream>>>(y_h, wP_t, b_proj, out, M, Cc, Cc);
  }
}
